// SmartContractGNN_49357764166283
// MI455X (gfx1250) — matrix operand order repaired and checked by reading, not compiled, not run on hardware
//
#include <hip/hip_runtime.h>
#include <hip/hip_bf16.h>

// ---------------- types ----------------
typedef __attribute__((ext_vector_type(8)))  __bf16 v8bf;
typedef __attribute__((ext_vector_type(16))) __bf16 v16bf;
typedef __attribute__((ext_vector_type(8)))  float  v8f;

#define D_IN   786
#define KP1    800   // 786 padded to 25*32
#define HCH    128
#define NHEADS 4
#define W1N    512   // HEADS*HCH

// ---------------- helpers ----------------
__device__ __forceinline__ float wsum(float v) {
#pragma unroll
    for (int o = 16; o > 0; o >>= 1) v += __shfl_xor(v, o, 32);
    return v;
}

__device__ __forceinline__ void atomicMaxF(float* addr, float value) {
    if (value >= 0.0f) {
        atomicMax((int*)addr, __float_as_int(value));
    } else {
        atomicMin((unsigned int*)addr, __float_as_uint(value));
    }
}

// ---------------- fill ----------------
__global__ __launch_bounds__(256) void k_fill(float* __restrict__ p, long long n, float v) {
    long long i = blockIdx.x * (long long)blockDim.x + threadIdx.x;
    if (i < n) p[i] = v;
}

// ---------------- weight convert: W[KxN] f32 -> Wt[N x Kp] bf16 (zero-padded K) ----------
__global__ __launch_bounds__(256)
void k_wt_bf16_t(const float* __restrict__ W, __bf16* __restrict__ Wt, int K, int N, int Kp) {
    long long idx = blockIdx.x * (long long)blockDim.x + threadIdx.x;
    long long total = (long long)N * Kp;
    if (idx >= total) return;
    int n = (int)(idx / Kp);
    int k = (int)(idx % Kp);
    Wt[idx] = (__bf16)(k < K ? W[(size_t)k * N + n] : 0.0f);
}

// ---------------- rows -> bf16 with optional layernorm (wave per row) ----------------
__global__ __launch_bounds__(256)
void k_rows_bf16(const float* __restrict__ x, const float* __restrict__ g,
                 const float* __restrict__ b, __bf16* __restrict__ out,
                 int Mreal, int Mpad, int K, int Kp, int do_ln) {
    int lane = threadIdx.x & 31;
    int row  = blockIdx.x * (blockDim.x >> 5) + (threadIdx.x >> 5);
    if (row >= Mpad) return;
    __bf16* orow = out + (size_t)row * Kp;
    if (row >= Mreal) {
        for (int k = lane; k < Kp; k += 32) orow[k] = (__bf16)0.0f;
        return;
    }
    const float* xr = x + (size_t)row * K;
    if (!do_ln) {
        for (int k = lane; k < Kp; k += 32) orow[k] = (__bf16)(k < K ? xr[k] : 0.0f);
        return;
    }
    float s = 0.0f, ss = 0.0f;
    for (int k = lane; k < K; k += 32) { float v = xr[k]; s += v; ss += v * v; }
    s = wsum(s); ss = wsum(ss);
    float mu  = s / (float)K;
    float var = ss / (float)K - mu * mu;
    float rs  = rsqrtf(var + 1e-5f);
    for (int k = lane; k < Kp; k += 32) {
        float v = 0.0f;
        if (k < K) v = (xr[k] - mu) * rs * g[k] + b[k];
        orow[k] = (__bf16)v;
    }
}

// ---------------- bf16 WMMA GEMM: C[Mreal x N] = A[Mpad x Kp] * Bt[N x Kp]^T (+bias) ------
// wave computes 16x64 tile via 4 accumulators of v_wmma_f32_16x16x32_bf16
__global__ __launch_bounds__(256)
void k_gemm_bf16(const __bf16* __restrict__ A, const __bf16* __restrict__ Bt,
                 const float* __restrict__ bias, float* __restrict__ C,
                 int Mreal, int Mpad, int N, int Kp) {
    const int lane = threadIdx.x & 31;
    const int wid  = blockIdx.x * (blockDim.x >> 5) + (threadIdx.x >> 5);
    const int ngroups = N >> 6;
    const int mtile = wid / ngroups;
    const int ngrp  = wid % ngroups;
    const int m0 = mtile * 16;
    if (m0 >= Mpad) return;
    const int n0 = ngrp * 64;

    const int half = lane >> 4;   // 0 | 1
    const int l15  = lane & 15;

    // A fragment base: row m0+l15, elements = K k0+half*16 .. +15
    const __bf16* ap  = A  + (size_t)(m0 + l15) * Kp + (half << 4);
    // B fragments: column n = n0+sub*16+l15 (row of Bt), elements = K k0+half*16 .. +15
    const __bf16* b0p = Bt + (size_t)(n0 +  0 + l15) * Kp + (half << 4);
    const __bf16* b1p = Bt + (size_t)(n0 + 16 + l15) * Kp + (half << 4);
    const __bf16* b2p = Bt + (size_t)(n0 + 32 + l15) * Kp + (half << 4);
    const __bf16* b3p = Bt + (size_t)(n0 + 48 + l15) * Kp + (half << 4);

    v8f acc[4] = {};
    for (int k0 = 0; k0 < Kp; k0 += 32) {
        union { v16bf v; v8bf h[2]; } af;
        af.h[0] = *(const v8bf*)(ap + k0);
        af.h[1] = *(const v8bf*)(ap + k0 + 8);
        v16bf bf0 = *(const v16bf*)(b0p + k0);
        v16bf bf1 = *(const v16bf*)(b1p + k0);
        v16bf bf2 = *(const v16bf*)(b2p + k0);
        v16bf bf3 = *(const v16bf*)(b3p + k0);
        if (k0 + 32 < Kp) {
            __builtin_prefetch(ap + k0 + 32, 0, 1);
            __builtin_prefetch(b0p + k0 + 32, 0, 1);
        }
        acc[0] = __builtin_amdgcn_wmma_f32_16x16x32_bf16(false, af.v, false, bf0, (short)0, acc[0], false, false);
        acc[1] = __builtin_amdgcn_wmma_f32_16x16x32_bf16(false, af.v, false, bf1, (short)0, acc[1], false, false);
        acc[2] = __builtin_amdgcn_wmma_f32_16x16x32_bf16(false, af.v, false, bf2, (short)0, acc[2], false, false);
        acc[3] = __builtin_amdgcn_wmma_f32_16x16x32_bf16(false, af.v, false, bf3, (short)0, acc[3], false, false);
    }

    float bsv[4];
#pragma unroll
    for (int j = 0; j < 4; ++j) bsv[j] = bias ? bias[n0 + j * 16 + l15] : 0.0f;

#pragma unroll
    for (int v = 0; v < 8; ++v) {
        int row = m0 + (half << 3) + v;   // D layout: vgpr v -> M = v + 8*half
        if (row < Mreal) {
            float* crow = C + (size_t)row * N + n0 + l15;
            crow[0]  = acc[0][v] + bsv[0];
            crow[16] = acc[1][v] + bsv[1];
            crow[32] = acc[2][v] + bsv[2];
            crow[48] = acc[3][v] + bsv[3];
        }
    }
}

// ---------------- per-node attention logits (wave per node*head) ----------------
__global__ __launch_bounds__(256)
void k_node_att(const float* __restrict__ h, const float* __restrict__ att_s,
                const float* __restrict__ att_d, float* __restrict__ as_,
                float* __restrict__ ad_, int Nn, int heads, int ch) {
    int lane = threadIdx.x & 31;
    int wid  = blockIdx.x * (blockDim.x >> 5) + (threadIdx.x >> 5);
    if (wid >= Nn * heads) return;
    int n = wid / heads, hd = wid % heads;
    const float* hp  = h + ((size_t)n * heads + hd) * ch;
    const float* asp = att_s + (size_t)hd * ch;
    const float* adp = att_d + (size_t)hd * ch;
    float ss = 0.0f, sd = 0.0f;
    for (int c = lane; c < ch; c += 32) { float v = hp[c]; ss += v * asp[c]; sd += v * adp[c]; }
    ss = wsum(ss); sd = wsum(sd);
    if (lane == 0) { as_[wid] = ss; ad_[wid] = sd; }
}

// ---------------- edge pass 1: leaky-relu score + segment max ----------------
__global__ __launch_bounds__(256)
void k_edge_score_max(const int* __restrict__ src, const int* __restrict__ dst,
                      const float* __restrict__ as_, const float* __restrict__ ad_,
                      float* __restrict__ ebuf, float* __restrict__ mbuf,
                      int E, int Nn, int heads) {
    long long idx = blockIdx.x * (long long)blockDim.x + threadIdx.x;
    long long total = (long long)(E + Nn) * heads;
    if (idx >= total) return;
    int e  = (int)(idx / heads);
    int hd = (int)(idx % heads);
    int s = (e < E) ? src[e] : (e - E);
    int d = (e < E) ? dst[e] : (e - E);
    float v = as_[s * heads + hd] + ad_[d * heads + hd];
    v = (v > 0.0f) ? v : 0.2f * v;
    ebuf[idx] = v;
    atomicMaxF(&mbuf[d * heads + hd], v);
}

// ---------------- edge pass 2: exp + segment sum ----------------
__global__ __launch_bounds__(256)
void k_edge_expsum(const int* __restrict__ src, const int* __restrict__ dst,
                   float* __restrict__ ebuf, const float* __restrict__ mbuf,
                   float* __restrict__ den, int E, int Nn, int heads) {
    long long idx = blockIdx.x * (long long)blockDim.x + threadIdx.x;
    long long total = (long long)(E + Nn) * heads;
    if (idx >= total) return;
    int e  = (int)(idx / heads);
    int hd = (int)(idx % heads);
    int d = (e < E) ? dst[e] : (e - E);
    float ex = __expf(ebuf[idx] - mbuf[d * heads + hd]);
    ebuf[idx] = ex;
    atomicAdd(&den[d * heads + hd], ex);
}

// ---------------- edge pass 3: weighted scatter-add aggregation (wave per edge*head) -----
__global__ __launch_bounds__(256)
void k_edge_agg(const int* __restrict__ src, const int* __restrict__ dst,
                const float* __restrict__ h, const float* __restrict__ ebuf,
                const float* __restrict__ den, float* __restrict__ out,
                int E, int Nn, int heads) {  // ch fixed = 128 (one float4 per lane)
    int lane = threadIdx.x & 31;
    long long wid = blockIdx.x * (long long)(blockDim.x >> 5) + (threadIdx.x >> 5);
    long long total = (long long)(E + Nn) * heads;
    if (wid >= total) return;
    int e  = (int)(wid / heads);
    int hd = (int)(wid % heads);
    int s = (e < E) ? src[e] : (e - E);
    int d = (e < E) ? dst[e] : (e - E);
    float alpha = ebuf[wid] / (den[d * heads + hd] + 1e-16f);
    const float4 v = ((const float4*)(h + ((size_t)s * heads + hd) * HCH))[lane];
    float* op = out + ((size_t)d * heads + hd) * HCH + lane * 4;
    atomicAdd(op + 0, v.x * alpha);
    atomicAdd(op + 1, v.y * alpha);
    atomicAdd(op + 2, v.z * alpha);
    atomicAdd(op + 3, v.w * alpha);
}

// ---------------- layer-1 finish: +bias, ELU, layernorm -> bf16 (wave per row, W=512) ----
__global__ __launch_bounds__(256)
void k_finish1(const float* __restrict__ agg, const float* __restrict__ bias,
               const float* __restrict__ g, const float* __restrict__ b,
               __bf16* __restrict__ out, int Mreal, int Mpad) {
    int lane = threadIdx.x & 31;
    int row  = blockIdx.x * (blockDim.x >> 5) + (threadIdx.x >> 5);
    if (row >= Mpad) return;
    __bf16* orow = out + (size_t)row * W1N;
    int c0 = lane * 16;
    if (row >= Mreal) {
        v16bf z;
#pragma unroll
        for (int i = 0; i < 16; ++i) z[i] = (__bf16)0.0f;
        *(v16bf*)(orow + c0) = z;
        return;
    }
    const float* ar = agg + (size_t)row * W1N;
    float v[16];
    float s = 0.0f, ss = 0.0f;
#pragma unroll
    for (int i = 0; i < 16; ++i) {
        float t = ar[c0 + i] + bias[c0 + i];
        t = (t > 0.0f) ? t : (__expf(t) - 1.0f);   // ELU
        v[i] = t; s += t; ss += t * t;
    }
    s = wsum(s); ss = wsum(ss);
    float mu  = s / (float)W1N;
    float var = ss / (float)W1N - mu * mu;
    float rs  = rsqrtf(var + 1e-5f);
    v16bf o;
#pragma unroll
    for (int i = 0; i < 16; ++i)
        o[i] = (__bf16)((v[i] - mu) * rs * g[c0 + i] + b[c0 + i]);
    *(v16bf*)(orow + c0) = o;
}

// ---------------- final: +bias2, LN, +identite, ELU, @Wc+bc (wave per node, ch=128) ------
__global__ __launch_bounds__(256)
void k_final(const float* __restrict__ agg, const float* __restrict__ bias2,
             const float* __restrict__ g2, const float* __restrict__ b2,
             const float* __restrict__ ident, const float* __restrict__ Wc,
             const float* __restrict__ bc, float* __restrict__ out, int Nn) {
    int lane = threadIdx.x & 31;
    int n = blockIdx.x * (blockDim.x >> 5) + (threadIdx.x >> 5);
    if (n >= Nn) return;
    const float* ar  = agg   + (size_t)n * HCH;
    const float* idr = ident + (size_t)n * HCH;
    int c0 = lane * 4;
    float v[4];
    float s = 0.0f, ss = 0.0f;
#pragma unroll
    for (int j = 0; j < 4; ++j) {
        float t = ar[c0 + j] + bias2[c0 + j];
        v[j] = t; s += t; ss += t * t;
    }
    s = wsum(s); ss = wsum(ss);
    float mu  = s / (float)HCH;
    float var = ss / (float)HCH - mu * mu;
    float rs  = rsqrtf(var + 1e-5f);
    float p0 = 0.0f, p1 = 0.0f;
#pragma unroll
    for (int j = 0; j < 4; ++j) {
        float t = (v[j] - mu) * rs * g2[c0 + j] + b2[c0 + j] + idr[c0 + j];
        t = (t > 0.0f) ? t : (__expf(t) - 1.0f);   // ELU
        p0 += t * Wc[(c0 + j) * 2 + 0];
        p1 += t * Wc[(c0 + j) * 2 + 1];
    }
    p0 = wsum(p0); p1 = wsum(p1);
    if (lane == 0) {
        out[(size_t)n * 2 + 0] = p0 + bc[0];
        out[(size_t)n * 2 + 1] = p1 + bc[1];
    }
}

// ======================= host launcher =======================
static inline int blocks_for_threads(long long t) { return (int)((t + 255) / 256); }
static inline int blocks_for_waves(long long w)   { return (int)((w + 7) / 8); }

extern "C" void kernel_launch(void* const* d_in, const int* in_sizes, int n_in,
                              void* d_out, int out_size, void* d_ws, size_t ws_size,
                              hipStream_t stream) {
    const float* x      = (const float*)d_in[0];
    const int*   ei     = (const int*)  d_in[1];
    const float* g_in   = (const float*)d_in[2];
    const float* b_in   = (const float*)d_in[3];
    const float* W1     = (const float*)d_in[4];
    const float* att_s1 = (const float*)d_in[5];
    const float* att_d1 = (const float*)d_in[6];
    const float* bias1  = (const float*)d_in[7];
    const float* g1     = (const float*)d_in[8];
    const float* b1     = (const float*)d_in[9];
    const float* W2     = (const float*)d_in[10];
    const float* att_s2 = (const float*)d_in[11];
    const float* att_d2 = (const float*)d_in[12];
    const float* bias2  = (const float*)d_in[13];
    const float* g2     = (const float*)d_in[14];
    const float* b2     = (const float*)d_in[15];
    const float* Wm     = (const float*)d_in[16];
    const float* bm     = (const float*)d_in[17];
    const float* Wc     = (const float*)d_in[18];
    const float* bc     = (const float*)d_in[19];
    float* outp = (float*)d_out;

    const int Nn = in_sizes[0] / D_IN;     // 50000
    const int E  = in_sizes[1] / 2;        // 800000
    const int Mpad = ((Nn + 15) / 16) * 16;
    const int* src = ei;
    const int* dst = ei + E;

    // -------- workspace layout --------
    size_t off = 0;
    auto alloc = [&](size_t bytes) { size_t o = off; off += (bytes + 255) & ~(size_t)255; return o; };
    char* ws = (char*)d_ws;

    size_t abuf_off  = alloc((size_t)Mpad * KP1 * 2);        // bf16 activations (reused: x, ln(x), then layer-2 input)
    size_t h1_off    = alloc((size_t)Nn * W1N * 4);          // f32 h1; reused for h2 + out2 in layer 2
    size_t out1_off  = alloc((size_t)Nn * W1N * 4);
    size_t ident_off = alloc((size_t)Nn * HCH * 4);
    size_t w1t_off   = alloc((size_t)W1N * KP1 * 2);
    size_t wmt_off   = alloc((size_t)HCH * KP1 * 2);
    size_t w2t_off   = alloc((size_t)HCH * W1N * 2);
    size_t as1_off   = alloc((size_t)Nn * NHEADS * 4);
    size_t ad1_off   = alloc((size_t)Nn * NHEADS * 4);
    size_t m1_off    = alloc((size_t)Nn * NHEADS * 4);
    size_t den1_off  = alloc((size_t)Nn * NHEADS * 4);
    size_t as2_off   = alloc((size_t)Nn * 4);
    size_t ad2_off   = alloc((size_t)Nn * 4);
    size_t m2_off    = alloc((size_t)Nn * 4);
    size_t den2_off  = alloc((size_t)Nn * 4);
    size_t ebuf_off  = alloc((size_t)(E + Nn) * NHEADS * 4);

    __bf16* Abf   = (__bf16*)(ws + abuf_off);   // [Mpad x 800] then [Mpad x 512]
    float*  h1f   = (float*) (ws + h1_off);
    float*  out1  = (float*) (ws + out1_off);
    float*  ident = (float*) (ws + ident_off);
    __bf16* W1t   = (__bf16*)(ws + w1t_off);
    __bf16* Wmt   = (__bf16*)(ws + wmt_off);
    __bf16* W2t   = (__bf16*)(ws + w2t_off);
    float*  as1   = (float*) (ws + as1_off);
    float*  ad1   = (float*) (ws + ad1_off);
    float*  m1    = (float*) (ws + m1_off);
    float*  den1  = (float*) (ws + den1_off);
    float*  as2   = (float*) (ws + as2_off);
    float*  ad2   = (float*) (ws + ad2_off);
    float*  m2    = (float*) (ws + m2_off);
    float*  den2  = (float*) (ws + den2_off);
    float*  ebuf  = (float*) (ws + ebuf_off);
    // layer-2 reuse inside h1f region
    float*  h2f   = h1f;                                    // [Nn x 128]
    float*  out2  = h1f + ((size_t)Nn * HCH + 64);          // [Nn x 128], disjoint (fits: Nn*512 total)

    // -------- weight conversion (bf16, transposed, K padded) --------
    k_wt_bf16_t<<<blocks_for_threads((long long)W1N * KP1), 256, 0, stream>>>(W1, W1t, D_IN, W1N, KP1);
    k_wt_bf16_t<<<blocks_for_threads((long long)HCH * KP1), 256, 0, stream>>>(Wm, Wmt, D_IN, HCH, KP1);
    k_wt_bf16_t<<<blocks_for_threads((long long)HCH * W1N), 256, 0, stream>>>(W2, W2t, W1N, HCH, W1N);

    // -------- residual: identite = x @ Wm + bm --------
    k_rows_bf16<<<blocks_for_waves(Mpad), 256, 0, stream>>>(x, nullptr, nullptr, Abf, Nn, Mpad, D_IN, KP1, 0);
    {
        long long waves = (long long)(Mpad / 16) * (HCH / 64);
        k_gemm_bf16<<<blocks_for_waves(waves), 256, 0, stream>>>(Abf, Wmt, bm, ident, Nn, Mpad, HCH, KP1);
    }

    // -------- h1 = ln(x) @ W1 --------
    k_rows_bf16<<<blocks_for_waves(Mpad), 256, 0, stream>>>(x, g_in, b_in, Abf, Nn, Mpad, D_IN, KP1, 1);
    {
        long long waves = (long long)(Mpad / 16) * (W1N / 64);
        k_gemm_bf16<<<blocks_for_waves(waves), 256, 0, stream>>>(Abf, W1t, nullptr, h1f, Nn, Mpad, W1N, KP1);
    }

    // -------- GAT layer 1 --------
    k_node_att<<<blocks_for_waves((long long)Nn * NHEADS), 256, 0, stream>>>(h1f, att_s1, att_d1, as1, ad1, Nn, NHEADS, HCH);
    k_fill<<<blocks_for_threads((long long)Nn * NHEADS), 256, 0, stream>>>(m1, (long long)Nn * NHEADS, -3.0e38f);
    k_fill<<<blocks_for_threads((long long)Nn * NHEADS), 256, 0, stream>>>(den1, (long long)Nn * NHEADS, 0.0f);
    k_fill<<<blocks_for_threads((long long)Nn * W1N), 256, 0, stream>>>(out1, (long long)Nn * W1N, 0.0f);
    {
        long long nEH = (long long)(E + Nn) * NHEADS;
        k_edge_score_max<<<blocks_for_threads(nEH), 256, 0, stream>>>(src, dst, as1, ad1, ebuf, m1, E, Nn, NHEADS);
        k_edge_expsum  <<<blocks_for_threads(nEH), 256, 0, stream>>>(src, dst, ebuf, m1, den1, E, Nn, NHEADS);
        k_edge_agg     <<<blocks_for_waves(nEH),   256, 0, stream>>>(src, dst, h1f, ebuf, den1, out1, E, Nn, NHEADS);
    }
    // +bias1, ELU, layernorm -> bf16 layer-2 input (reuses Abf, now [Mpad x 512])
    k_finish1<<<blocks_for_waves(Mpad), 256, 0, stream>>>(out1, bias1, g1, b1, Abf, Nn, Mpad);

    // -------- h2 = A2 @ W2 --------
    {
        long long waves = (long long)(Mpad / 16) * (HCH / 64);
        k_gemm_bf16<<<blocks_for_waves(waves), 256, 0, stream>>>(Abf, W2t, nullptr, h2f, Nn, Mpad, HCH, W1N);
    }

    // -------- GAT layer 2 (heads = 1) --------
    k_node_att<<<blocks_for_waves((long long)Nn), 256, 0, stream>>>(h2f, att_s2, att_d2, as2, ad2, Nn, 1, HCH);
    k_fill<<<blocks_for_threads((long long)Nn), 256, 0, stream>>>(m2, (long long)Nn, -3.0e38f);
    k_fill<<<blocks_for_threads((long long)Nn), 256, 0, stream>>>(den2, (long long)Nn, 0.0f);
    k_fill<<<blocks_for_threads((long long)Nn * HCH), 256, 0, stream>>>(out2, (long long)Nn * HCH, 0.0f);
    {
        long long nEH = (long long)(E + Nn);
        k_edge_score_max<<<blocks_for_threads(nEH), 256, 0, stream>>>(src, dst, as2, ad2, ebuf, m2, E, Nn, 1);
        k_edge_expsum  <<<blocks_for_threads(nEH), 256, 0, stream>>>(src, dst, ebuf, m2, den2, E, Nn, 1);
        k_edge_agg     <<<blocks_for_waves(nEH),   256, 0, stream>>>(src, dst, h2f, ebuf, den2, out2, E, Nn, 1);
    }

    // -------- final: +bias2, LN(g2,b2), +identite, ELU, @Wc + bc --------
    k_final<<<blocks_for_waves((long long)Nn), 256, 0, stream>>>(out2, bias2, g2, b2, ident, Wc, bc, outp, Nn);
}
